// LocMotionAppearanceGCN_56959856279909
// MI455X (gfx1250) — matrix operand order repaired and checked by reading, not compiled, not run on hardware
//
#include <hip/hip_runtime.h>
#include <hip/hip_bf16.h>

// ---------------------------------------------------------------------------
// Types for CDNA5 WMMA (wave32, 16x16x32 bf16 -> f32)
// ---------------------------------------------------------------------------
typedef __attribute__((ext_vector_type(16))) __bf16        v16bf;
typedef __attribute__((ext_vector_type(8)))  float         v8f;
typedef __attribute__((ext_vector_type(8)))  unsigned short ushort8;

#define NSP   1500
#define NNODE 6000
#define CDIM  256
#define BIMG  4
#define HIMG  384
#define WIMG  384
#define HFEAT 96
#define WFEAT 96

__device__ __forceinline__ unsigned short f2bf(float f) {
    unsigned int u = __float_as_uint(f);
    unsigned int r = u + 0x7FFFu + ((u >> 16) & 1u);   // round-to-nearest-even
    return (unsigned short)(r >> 16);
}

// ---------------------------------------------------------------------------
// generic fill
// ---------------------------------------------------------------------------
__global__ void fill_f32(float* __restrict__ p, float v, int n) {
    int i = blockIdx.x * blockDim.x + threadIdx.x;
    if (i < n) p[i] = v;
}

// ---------------------------------------------------------------------------
// Edge weights + degree accumulation.  deg pre-initialized to 2.0 (self loops)
// ---------------------------------------------------------------------------
__global__ void edge_w_deg(const int* __restrict__ edges, const float* __restrict__ probas,
                           float* __restrict__ wbuf, float* __restrict__ deg, int E) {
    int e = blockIdx.x * blockDim.x + threadIdx.x;
    if (e >= E) return;
    int s = edges[e], d = edges[E + e];
    float w = __expf(-fabsf(probas[s] - probas[d]) * 5.0f);   // 1/SIGMA = 5
    wbuf[e] = w;
    atomicAdd(&deg[d], w);
}

__global__ void deg_to_dis(float* __restrict__ deg, int n) {
    int i = blockIdx.x * blockDim.x + threadIdx.x;
    if (i < n) {
        float dg = deg[i];                      // >= 2, so always > 0
        deg[i] = rsqrtf(fmaxf(dg, 1e-30f));
    }
}

__global__ void edge_norm(float* __restrict__ wbuf, const float* __restrict__ dis,
                          const int* __restrict__ edges, int E) {
    int e = blockIdx.x * blockDim.x + threadIdx.x;
    if (e >= E) return;
    int s = edges[e], d = edges[E + e];
    wbuf[e] = dis[s] * wbuf[e] * dis[d];
}

// ---------------------------------------------------------------------------
// Superpixel pooling of the raw image (3 channels) + pixel counts
// ---------------------------------------------------------------------------
__global__ void pool_image(const float* __restrict__ img, const int* __restrict__ labels,
                           float* __restrict__ zsum, float* __restrict__ cnt) {
    int by = blockIdx.x;              // B*H rows
    int b = by / HIMG, y = by % HIMG;
    int x = threadIdx.x;              // 384 threads
    int seg = labels[(b * HIMG + y) * WIMG + x] + b * NSP;
    const float* p = img + ((size_t)b * 3 * HIMG * WIMG) + (size_t)y * WIMG + x;
    atomicAdd(&zsum[seg * 3 + 0], p[0]);
    atomicAdd(&zsum[seg * 3 + 1], p[(size_t)HIMG * WIMG]);
    atomicAdd(&zsum[seg * 3 + 2], p[2 * (size_t)HIMG * WIMG]);
    atomicAdd(&cnt[seg], 1.0f);
}

__global__ void div3(float* __restrict__ zsum, const float* __restrict__ cnt, int n) {
    int i = blockIdx.x * blockDim.x + threadIdx.x;
    if (i >= n) return;
    float inv = 1.0f / fmaxf(cnt[i], 1.0f);
    zsum[i * 3 + 0] *= inv;
    zsum[i * 3 + 1] *= inv;
    zsum[i * 3 + 2] *= inv;
}

// ---------------------------------------------------------------------------
// Fused bilinear-upsample (96->384, align-corners linspace) + superpixel sum.
// One block per (b,y) output row, one thread per channel; run-length
// compression of the per-pixel atomics along x (superpixels are contiguous).
// ---------------------------------------------------------------------------
__global__ void pool_feats(const float* __restrict__ f, const int* __restrict__ labels,
                           float* __restrict__ hsum) {
    const float SC = 95.0f / 383.0f;
    int by = blockIdx.x;              // B*H
    int b = by / HIMG, y = by % HIMG;
    int c = threadIdx.x;              // 256 channels

    float ysf = y * SC;
    int y0 = (int)ysf;
    int y1 = min(y0 + 1, HFEAT - 1);
    float wy = ysf - (float)y0;

    __shared__ int   sseg[WIMG];
    __shared__ int   sx0[WIMG];
    __shared__ float swx[WIMG];
    const int* lab = labels + (b * HIMG + y) * WIMG;
    for (int x = c; x < WIMG; x += blockDim.x) {
        sseg[x] = lab[x] + b * NSP;
        float xsf = x * SC;
        int x0 = (int)xsf;
        sx0[x] = x0;
        swx[x] = xsf - (float)x0;
    }
    __syncthreads();

    const float* f0 = f + (((size_t)b * CDIM + c) * HFEAT + y0) * WFEAT;
    const float* f1 = f + (((size_t)b * CDIM + c) * HFEAT + y1) * WFEAT;

    int   cur = sseg[0];
    float acc = 0.0f;
    for (int x = 0; x < WIMG; ++x) {
        int   x0 = sx0[x];
        int   x1 = min(x0 + 1, WFEAT - 1);
        float wx = swx[x];
        float top = f0[x0] * (1.0f - wx) + f0[x1] * wx;
        float bot = f1[x0] * (1.0f - wx) + f1[x1] * wx;
        float v   = top * (1.0f - wy) + bot * wy;
        int sg = sseg[x];
        if (sg != cur) {
            atomicAdd(&hsum[(size_t)cur * CDIM + c], acc);
            cur = sg; acc = v;
        } else {
            acc += v;
        }
    }
    atomicAdd(&hsum[(size_t)cur * CDIM + c], acc);
}

// ---------------------------------------------------------------------------
// Layer 0 projection: xw = Z3(6000x3) @ W0(3x256)  (K=3: plain VALU)
// ---------------------------------------------------------------------------
__global__ void gcn0_xw(const float* __restrict__ z3, const float* __restrict__ w0,
                        float* __restrict__ xw) {
    int idx = blockIdx.x * blockDim.x + threadIdx.x;  // 6000*256
    int i = idx >> 8, j = idx & 255;
    xw[idx] = z3[i * 3 + 0] * w0[j] +
              z3[i * 3 + 1] * w0[256 + j] +
              z3[i * 3 + 2] * w0[512 + j];
}

// ---------------------------------------------------------------------------
// Mix (0.5*Hp/cnt + 0.5*Z) and cast to bf16 GEMM A operand
// ---------------------------------------------------------------------------
__global__ void mix_cast(const float* __restrict__ hsum, const float* __restrict__ cnt,
                         const float* __restrict__ z, unsigned short* __restrict__ zbf) {
    int idx = blockIdx.x * blockDim.x + threadIdx.x;
    int i = idx >> 8;
    float hp = hsum[idx] / fmaxf(cnt[i], 1.0f);
    zbf[idx] = f2bf(0.5f * hp + 0.5f * z[idx]);
}

// ---------------------------------------------------------------------------
// Pack W (256x256 row-major, rows=K) into the wave32 WMMA B-operand layout:
// flat = ((kt*16 + nt)*32 + lane)*16 + e ;  K = kt*32 + (e>>3)*16 + (lane>>4)*8 + (e&7),
// N = nt*16 + (lane&15).  Then the GEMM loads 32 contiguous bytes per lane.
// ---------------------------------------------------------------------------
__global__ void pack_w(const float* __restrict__ w, unsigned short* __restrict__ wp) {
    int idx = blockIdx.x * blockDim.x + threadIdx.x;  // 65536
    int e    = idx & 15;
    int lane = (idx >> 4) & 31;
    int nt   = (idx >> 9) & 15;
    int kt   = idx >> 13;
    int k = kt * 32 + (e >> 3) * 16 + (lane >> 4) * 8 + (e & 7);
    int n = nt * 16 + (lane & 15);
    wp[idx] = f2bf(w[k * 256 + n]);
}

// ---------------------------------------------------------------------------
// WMMA GEMM: xw(6000x256,f32) = Zbf(6000x256,bf16) @ Wpack(256x256,bf16)
// 8 waves/block, one 16x16 output tile per wave, 8 chained v_wmma per tile.
// ---------------------------------------------------------------------------
__global__ void wmma_gemm(const unsigned short* __restrict__ zbf,
                          const unsigned short* __restrict__ wp,
                          float* __restrict__ xw) {
    int wave = threadIdx.x >> 5;
    int lane = threadIdx.x & 31;
    int tile = blockIdx.x * 8 + wave;          // 375*16 = 6000 tiles
    int mt = tile >> 4, nt = tile & 15;
    int m    = lane & 15;
    int half = lane >> 4;

    const unsigned short* arow = zbf + (size_t)(mt * 16 + m) * CDIM;
    v8f acc = {};

    union AU { v16bf v; ushort8 h[2]; };
    union BU { v16bf v; ushort8 h[2]; };

    #pragma unroll
    for (int kt = 0; kt < 8; ++kt) {
        AU a;
        // 16-bit A 16x32 layout: half 0 -> K 0..7 / 16..23, half 1 -> 8..15 / 24..31
        a.h[0] = *(const ushort8*)(arow + kt * 32 + half * 8);
        a.h[1] = *(const ushort8*)(arow + kt * 32 + 16 + half * 8);
        BU bm;
        const unsigned short* bp = wp + (((size_t)(kt * 16 + nt) * 32 + lane) * 16);
        bm.h[0] = *(const ushort8*)(bp);
        bm.h[1] = *(const ushort8*)(bp + 8);
        if (kt < 7) {
            __builtin_prefetch(arow + (kt + 1) * 32, 0, 0);   // global_prefetch_b8
        }
        acc = __builtin_amdgcn_wmma_f32_16x16x32_bf16(
                  false, a.v, false, bm.v, (short)0, acc, false, false);
    }

    // D layout: VGPR r -> M = half*8 + r, N = lane&15
    int mbase = mt * 16 + half * 8;
    int ncol  = nt * 16 + m;
    #pragma unroll
    for (int r = 0; r < 8; ++r)
        xw[(size_t)(mbase + r) * CDIM + ncol] = acc[r];
}

// ---------------------------------------------------------------------------
// out = bias + 2*dis^2*xw  (bias + both self-loop edges fused)
// ---------------------------------------------------------------------------
__global__ void init_out(float* __restrict__ out, const float* __restrict__ bias,
                         const float* __restrict__ dis, const float* __restrict__ xw) {
    int idx = blockIdx.x * blockDim.x + threadIdx.x;
    int i = idx >> 8, j = idx & 255;
    float di = dis[i];
    out[idx] = bias[j] + 2.0f * di * di * xw[idx];
}

// ---------------------------------------------------------------------------
// Edge scatter: out[dst] += norm_e * xw[src]  (256 channels per edge)
// ---------------------------------------------------------------------------
__global__ void edge_scatter(float* __restrict__ out, const float* __restrict__ xw,
                             const float* __restrict__ ne, const int* __restrict__ edges,
                             int E, int epb) {
    int j  = threadIdx.x;
    int e0 = blockIdx.x * epb;
    for (int k = 0; k < epb; ++k) {
        int e = e0 + k;
        if (e >= E) return;
        int s = edges[e], d = edges[E + e];
        float nv = ne[e];
        atomicAdd(&out[(size_t)d * CDIM + j], nv * xw[(size_t)s * CDIM + j]);
    }
}

__global__ void relu_ip(float* __restrict__ z, int n) {
    int i = blockIdx.x * blockDim.x + threadIdx.x;
    if (i < n) z[i] = fmaxf(z[i], 0.0f);
}

// ---------------------------------------------------------------------------
// Row L2-normalize Z -> cs_r  (block = 256 threads = one row, wave32 reduce)
// ---------------------------------------------------------------------------
__global__ void row_norm(const float* __restrict__ z, float* __restrict__ csr) {
    int i = blockIdx.x;
    int j = threadIdx.x;
    float v  = z[(size_t)i * CDIM + j];
    float ss = v * v;
    #pragma unroll
    for (int o = 16; o > 0; o >>= 1) ss += __shfl_down(ss, o, 32);
    __shared__ float red[8];
    if ((j & 31) == 0) red[j >> 5] = ss;
    __syncthreads();
    float tot = red[0] + red[1] + red[2] + red[3] + red[4] + red[5] + red[6] + red[7];
    float inv = 1.0f / fmaxf(sqrtf(tot), 1e-12f);
    csr[(size_t)i * CDIM + j] = v * inv;
}

__global__ void wn_norm(const float* __restrict__ lw, float* __restrict__ wn) {
    int i = blockIdx.x;     // 15 rows
    int j = threadIdx.x;    // 256
    float v  = lw[(size_t)i * CDIM + j];
    float ss = v * v;
    #pragma unroll
    for (int o = 16; o > 0; o >>= 1) ss += __shfl_down(ss, o, 32);
    __shared__ float red[8];
    if ((j & 31) == 0) red[j >> 5] = ss;
    __syncthreads();
    float tot = red[0] + red[1] + red[2] + red[3] + red[4] + red[5] + red[6] + red[7];
    wn[(size_t)i * CDIM + j] = v / fmaxf(sqrtf(tot), 1e-30f);
}

__global__ void cs_proj(const float* __restrict__ csr, const float* __restrict__ wn,
                        float* __restrict__ cs, int n) {
    int idx = blockIdx.x * blockDim.x + threadIdx.x;  // 6000*15
    if (idx >= n) return;
    int i = idx / 15, t = idx % 15;
    const float* a = csr + (size_t)i * CDIM;
    const float* b = wn  + (size_t)t * CDIM;
    float s = 0.0f;
    #pragma unroll 8
    for (int j = 0; j < CDIM; ++j) s += a[j] * b[j];
    cs[idx] = s;
}

// ---------------------------------------------------------------------------
// Host launcher
// ---------------------------------------------------------------------------
extern "C" void kernel_launch(void* const* d_in, const int* in_sizes, int n_in,
                              void* d_out, int out_size, void* d_ws, size_t ws_size,
                              hipStream_t stream) {
    const float* image  = (const float*)d_in[0];
    const int*   labels = (const int*)  d_in[1];
    const int*   edges  = (const int*)  d_in[2];
    const float* probas = (const float*)d_in[3];
    const float* feats0 = (const float*)d_in[4];
    const float* feats1 = (const float*)d_in[5];
    const float* W0     = (const float*)d_in[6];
    const float* b0     = (const float*)d_in[7];
    const float* W1     = (const float*)d_in[8];
    const float* b1     = (const float*)d_in[9];
    const float* W2     = (const float*)d_in[10];
    const float* b2     = (const float*)d_in[11];
    const float* lin_w  = (const float*)d_in[12];

    const int E = in_sizes[2] / 2;        // 120000
    const int N = in_sizes[3];            // 6000
    const int NC = N * CDIM;              // 1,536,000

    // workspace carve-up (bytes)
    char* ws = (char*)d_ws;
    float*          Z    = (float*)(ws);                         // N*C f32
    float*          XW   = (float*)(ws + 6144000);               // N*C f32
    float*          HP   = (float*)(ws + 12288000);              // N*C f32
    unsigned short* ZBF  = (unsigned short*)(ws + 18432000);     // N*C bf16
    unsigned short* WP   = (unsigned short*)(ws + 21504000);     // 65536 bf16
    float*          WN   = (float*)(ws + 21635072);              // 15*256
    float*          DIS  = (float*)(ws + 21650432);              // N (deg->dis)
    float*          CNT  = (float*)(ws + 21674432);              // N
    float*          Z3   = (float*)(ws + 21698432);              // N*3
    float*          NE   = (float*)(ws + 21770432);              // E
    float*          CS   = (float*)d_out;                        // 6000*15
    float*          CSR  = (float*)d_out + N * 15;               // 6000*256

    const int T = 256;
    auto g = [](int n, int t) { return (n + t - 1) / t; };

    // ---- graph normalization -------------------------------------------
    fill_f32<<<g(N, T), T, 0, stream>>>(DIS, 2.0f, N);           // deg init = self loops
    fill_f32<<<g(N, T), T, 0, stream>>>(CNT, 0.0f, N);
    fill_f32<<<g(N * 3, T), T, 0, stream>>>(Z3, 0.0f, N * 3);
    edge_w_deg<<<g(E, T), T, 0, stream>>>(edges, probas, NE, DIS, E);
    deg_to_dis<<<g(N, T), T, 0, stream>>>(DIS, N);
    edge_norm<<<g(E, T), T, 0, stream>>>(NE, DIS, edges, E);

    // ---- layer 0: image pooling + K=3 projection -----------------------
    pool_image<<<BIMG * HIMG, WIMG, 0, stream>>>(image, labels, Z3, CNT);
    div3<<<g(N, T), T, 0, stream>>>(Z3, CNT, N);
    gcn0_xw<<<g(NC, T), T, 0, stream>>>(Z3, W0, XW);
    init_out<<<g(NC, T), T, 0, stream>>>(Z, b0, DIS, XW);
    edge_scatter<<<g(E, 8), T, 0, stream>>>(Z, XW, NE, edges, E, 8);
    relu_ip<<<g(NC, T), T, 0, stream>>>(Z, NC);

    // ---- layers 1,2: feats pooling + bf16 WMMA GEMM + scatter ----------
    const float* feats[2] = {feats0, feats1};
    const float* Wk[2]    = {W1, W2};
    const float* bk[2]    = {b1, b2};
    for (int k = 0; k < 2; ++k) {
        fill_f32<<<g(NC, T), T, 0, stream>>>(HP, 0.0f, NC);
        pool_feats<<<BIMG * HIMG, CDIM, 0, stream>>>(feats[k], labels, HP);
        mix_cast<<<g(NC, T), T, 0, stream>>>(HP, CNT, Z, ZBF);
        pack_w<<<g(65536, T), T, 0, stream>>>(Wk[k], WP);
        wmma_gemm<<<N * 16 / 8, T, 0, stream>>>(ZBF, WP, XW);    // 12000 blocks
        init_out<<<g(NC, T), T, 0, stream>>>(Z, bk[k], DIS, XW);
        edge_scatter<<<g(E, 8), T, 0, stream>>>(Z, XW, NE, edges, E, 8);
        relu_ip<<<g(NC, T), T, 0, stream>>>(Z, NC);
    }

    // ---- outputs --------------------------------------------------------
    row_norm<<<N, CDIM, 0, stream>>>(Z, CSR);
    wn_norm<<<15, CDIM, 0, stream>>>(lin_w, WN);
    cs_proj<<<g(N * 15, T), T, 0, stream>>>(CSR, WN, CS, N * 15);
}
